// CausalSelfAttention_83193516523861
// MI455X (gfx1250) — matrix operand order repaired and checked by reading, not compiled, not run on hardware
//
#include <hip/hip_runtime.h>
#include <hip/hip_bf16.h>

typedef __attribute__((ext_vector_type(16))) _Float16 v16h;
typedef __attribute__((ext_vector_type(8)))  _Float16 v8h;
typedef __attribute__((ext_vector_type(4)))  _Float16 v4h;
typedef __attribute__((ext_vector_type(8)))  float    v8f;
typedef __attribute__((ext_vector_type(4)))  unsigned int u32x4;
typedef __attribute__((ext_vector_type(8)))  int          i32x8;
typedef __attribute__((ext_vector_type(4)))  int          i32x4;

#define WMMA_F16(a, b, c) \
  __builtin_amdgcn_wmma_f32_16x16x32_f16(false, (a), false, (b), (short)0, (c), false, false)

#if defined(__AMDGCN__) && __has_builtin(__builtin_amdgcn_tensor_load_to_lds) && \
    __has_builtin(__builtin_amdgcn_s_wait_tensorcnt)
#define USE_TDM 1
#if __has_include(<hip/amd_detail/amd_gfx1250_TDM.h>)
#define TDM_ARITY6 1
#endif
#endif

constexpr int B_ = 4, T_ = 2048, C_ = 1024, H_ = 16, D_ = 64;
constexpr int BT    = B_ * T_;   // 8192 rows
constexpr int N_QKV = 3 * C_;    // 3072 qkv cols

// ---------------------------------------------------------------------------
// Fragment loaders (ISA 7.12.2 wave32 layouts, f16 16x16x32)
// A (16x32, MxK): lane L row M=L%16; elems 0..7 -> K=base0+0..7,
//                 elems 8..15 -> K=16+base0+0..7, base0=(L>=16)?8:0.
// B (32x16, KxN): lane L col N=L%16; elems 0..7 -> K=base0+0..7,
//                 elems 8..15 -> K=16+base0+0..7, base0=(L>=16)?8:0.
// ---------------------------------------------------------------------------
__device__ __forceinline__ v16h load_a_frag(const _Float16* p, int lane) {
  const int base0 = (lane >= 16) ? 8 : 0;
  v8h lo = *(const v8h*)(p + base0);
  v8h hi = *(const v8h*)(p + 16 + base0);
  v16h a;
#pragma unroll
  for (int i = 0; i < 8; ++i) { a[i] = lo[i]; a[i + 8] = hi[i]; }
  return a;
}

__device__ __forceinline__ v16h load_b_frag(const _Float16* p, int lane) {
  const int baseK = (lane >= 16) ? 8 : 0;
  v8h lo = *(const v8h*)(p + baseK);
  v8h hi = *(const v8h*)(p + baseK + 16);
  v16h b;
#pragma unroll
  for (int i = 0; i < 8; ++i) { b[i] = lo[i]; b[i + 8] = hi[i]; }
  return b;
}

// ---------------------------------------------------------------------------
// TDM: 2D tile load Global -> LDS (descriptor per ISA ch.8; data_size = 2B)
// ---------------------------------------------------------------------------
__device__ __forceinline__ void tdm_load_2d(const _Float16* gptr, unsigned lds_byte,
                                            unsigned tile_d0, unsigned tile_d1,
                                            unsigned long stride0_elems,
                                            unsigned tensor_d0, unsigned tensor_d1) {
#ifdef USE_TDM
  const unsigned long ga = (unsigned long)gptr;
  u32x4 g0;
  g0[0] = 1u;                                     // count=1, user descriptor
  g0[1] = lds_byte;                               // lds_addr (bytes)
  g0[2] = (unsigned)(ga & 0xffffffffu);           // global_addr[31:0]
  g0[3] = (unsigned)((ga >> 32) & 0x1ffffffu) | (2u << 30);  // addr[56:32] | type=2
  i32x8 g1;
  g1[0] = (int)(1u << 16);                        // wg_mask=0, data_size=1 (2 bytes)
  g1[1] = (int)((tensor_d0 & 0xffffu) << 16);     // abar=0 | tensor_dim0[15:0]
  g1[2] = (int)((tensor_d0 >> 16) | ((tensor_d1 & 0xffffu) << 16));
  g1[3] = (int)((tensor_d1 >> 16) | (tile_d0 << 16));
  g1[4] = (int)(tile_d1 & 0xffffu);               // tile_dim1 | tile_dim2=0
  g1[5] = (int)(stride0_elems & 0xffffffffu);     // tensor_dim0_stride[31:0]
  g1[6] = (int)((stride0_elems >> 32) & 0xffffu); // stride0[47:32] | stride1=0
  g1[7] = 0;
  i32x4 gz = {0, 0, 0, 0};
#ifdef TDM_ARITY6
  i32x8 g4 = {0, 0, 0, 0, 0, 0, 0, 0};
  __builtin_amdgcn_tensor_load_to_lds(g0, g1, gz, gz, g4, 0);
#else
  __builtin_amdgcn_tensor_load_to_lds(g0, g1, gz, gz, 0);
#endif
#else
  (void)gptr; (void)lds_byte; (void)tile_d0; (void)tile_d1;
  (void)stride0_elems; (void)tensor_d0; (void)tensor_d1;
#endif
}

// ---------------------------------------------------------------------------
// Converts
// ---------------------------------------------------------------------------
__global__ void cvt_f32_to_f16(const float* __restrict__ src,
                               _Float16* __restrict__ dst, int n4) {
  int i = blockIdx.x * blockDim.x + threadIdx.x;
  if (i >= n4) return;
  float4 v = ((const float4*)src)[i];
  v4h h;
  h[0] = (_Float16)v.x; h[1] = (_Float16)v.y;
  h[2] = (_Float16)v.z; h[3] = (_Float16)v.w;
  ((v4h*)dst)[i] = h;
}

// w [K][N] f32 -> wT [N][K] f16
__global__ void cvt_transpose(const float* __restrict__ w,
                              _Float16* __restrict__ wT, int K, int N) {
  long idx = blockIdx.x * (long)blockDim.x + threadIdx.x;
  if (idx >= (long)K * N) return;
  int n = (int)(idx / K), k = (int)(idx % K);
  wT[idx] = (_Float16)w[(long)k * N + n];
}

// ---------------------------------------------------------------------------
// Pipelined WMMA GEMM core: 32 rows x 64 cols per wave, ping-pong k-chunks
// ---------------------------------------------------------------------------
struct Frags {
  v16h a0, a1, b0, b1, b2, b3;
};

__device__ __forceinline__ Frags load_frags(const _Float16* pa, const _Float16* pb,
                                            int kk, int lane) {
  Frags f;
  f.a0 = load_a_frag(pa + kk, lane);
  f.a1 = load_a_frag(pa + 16 * C_ + kk, lane);
  f.b0 = load_b_frag(pb + kk, lane);
  f.b1 = load_b_frag(pb + 16 * C_ + kk, lane);
  f.b2 = load_b_frag(pb + 32 * C_ + kk, lane);
  f.b3 = load_b_frag(pb + 48 * C_ + kk, lane);
  return f;
}

__device__ __forceinline__ void do_wmma8(const Frags& f, v8f acc[2][4]) {
  acc[0][0] = WMMA_F16(f.a0, f.b0, acc[0][0]);
  acc[0][1] = WMMA_F16(f.a0, f.b1, acc[0][1]);
  acc[0][2] = WMMA_F16(f.a0, f.b2, acc[0][2]);
  acc[0][3] = WMMA_F16(f.a0, f.b3, acc[0][3]);
  acc[1][0] = WMMA_F16(f.a1, f.b0, acc[1][0]);
  acc[1][1] = WMMA_F16(f.a1, f.b1, acc[1][1]);
  acc[1][2] = WMMA_F16(f.a1, f.b2, acc[1][2]);
  acc[1][3] = WMMA_F16(f.a1, f.b3, acc[1][3]);
}

__device__ __forceinline__ void gemm_core(const _Float16* pa, const _Float16* pb,
                                          int lane, v8f acc[2][4]) {
  Frags f0 = load_frags(pa, pb, 0, lane);
  Frags f1 = load_frags(pa, pb, 32, lane);
#pragma unroll 1
  for (int kk = 0; kk < C_; kk += 64) {
    __builtin_prefetch(pa + kk + 256, 0, 1);
    __builtin_prefetch(pb + kk + 256, 0, 1);
    do_wmma8(f0, acc);
    if (kk + 64 < C_) f0 = load_frags(pa, pb, kk + 64, lane);
    do_wmma8(f1, acc);
    if (kk + 96 < C_) f1 = load_frags(pa, pb, kk + 96, lane);
  }
}

// ---------------------------------------------------------------------------
// QKV GEMM: [BT x C] @ [C x 3C] + bias -> Q,K [B,H,T,D] f16, V [B,H,D,T] f16
// ---------------------------------------------------------------------------
__global__ __launch_bounds__(256) void gemm_qkv(
    const _Float16* __restrict__ xh, const _Float16* __restrict__ wT,
    const float* __restrict__ bias, _Float16* __restrict__ qh,
    _Float16* __restrict__ kh, _Float16* __restrict__ vt) {
  const int lane = threadIdx.x & 31;
  const int wave = threadIdx.x >> 5;
  const int m0 = blockIdx.y * 32;
  const int n0 = (blockIdx.x * 8 + wave) * 64;
  const int cn = lane & 15;
  const int half8 = (lane >= 16) ? 8 : 0;

  v8f zero = {0.f, 0.f, 0.f, 0.f, 0.f, 0.f, 0.f, 0.f};
  v8f acc[2][4] = {{zero, zero, zero, zero}, {zero, zero, zero, zero}};

  gemm_core(xh + (long)(m0 + cn) * C_, wT + (long)(n0 + cn) * C_, lane, acc);

#pragma unroll
  for (int nt = 0; nt < 4; ++nt) {
    const int col = n0 + nt * 16 + cn;
    const float bv = bias[col];
    const int which = col >> 10;  // 0=q 1=k 2=v
    const int cm = col & (C_ - 1);
    const int h = cm >> 6, d = cm & 63;
#pragma unroll
    for (int mt = 0; mt < 2; ++mt) {
#pragma unroll
      for (int r = 0; r < 8; ++r) {
        const int row = m0 + mt * 16 + r + half8;
        const int bb = row >> 11, t = row & (T_ - 1);
        const long bh = (long)bb * H_ + h;
        const _Float16 hv = (_Float16)(acc[mt][nt][r] + bv);
        if (which == 0)      qh[(bh * T_ + t) * D_ + d] = hv;
        else if (which == 1) kh[(bh * T_ + t) * D_ + d] = hv;
        else                 vt[(bh * D_ + d) * T_ + t] = hv;
      }
    }
  }
}

// ---------------------------------------------------------------------------
// Causal flash attention: 1 wave per (b,h, 32-row query tile).
// TDM double-buffers 32x64 K-chunk and 64x32 V-chunk into LDS.
// ---------------------------------------------------------------------------
__global__ __launch_bounds__(32) void attn_fwd(
    const _Float16* __restrict__ qh, const _Float16* __restrict__ kh,
    const _Float16* __restrict__ vt, _Float16* __restrict__ yh) {
  // halves: K buf0 [0,2048) buf1 [2048,4096); V buf0 [4096,6144) buf1 [6144,8192);
  // P tile [8192, 9216)
  __shared__ __align__(16) _Float16 smem[9216];
  _Float16* const pt = smem + 8192;

  const int lane = threadIdx.x & 31;
  const int cn = lane & 15;
  const int half8 = (lane >= 16) ? 8 : 0;
  const int bh = blockIdx.x;
  const int t0 = blockIdx.y * 32;

  const _Float16* Q = qh + (long)bh * T_ * D_;
  const _Float16* K = kh + (long)bh * T_ * D_;
  const _Float16* V = vt + (long)bh * D_ * T_;

  v16h qf[2][2];
#pragma unroll
  for (int mt = 0; mt < 2; ++mt) {
    const _Float16* pq = Q + (long)(t0 + mt * 16 + cn) * D_;
    qf[mt][0] = load_a_frag(pq, lane);
    qf[mt][1] = load_a_frag(pq + 32, lane);
  }

  v8f zero = {0.f, 0.f, 0.f, 0.f, 0.f, 0.f, 0.f, 0.f};
  v8f o[2][4] = {{zero, zero, zero, zero}, {zero, zero, zero, zero}};
  float m[2][8], l[2][8];
#pragma unroll
  for (int mt = 0; mt < 2; ++mt)
#pragma unroll
    for (int r = 0; r < 8; ++r) { m[mt][r] = -1e30f; l[mt][r] = 0.f; }

  const int nch = t0 / 32 + 1;
#ifdef USE_TDM
  tdm_load_2d(K, 0u, 64, 32, 64, 64, T_);
  tdm_load_2d(V, 8192u, 32, 64, T_, T_, D_);
#endif

  for (int ci = 0; ci < nch; ++ci) {
    const int s0 = ci * 32;
    const int buf = ci & 1;
#ifdef USE_TDM
    if (ci + 1 < nch) {
      const unsigned nb = (unsigned)(buf ^ 1);
      tdm_load_2d(K + (long)(s0 + 32) * D_, nb * 4096u, 64, 32, 64, 64, T_);
      tdm_load_2d(V + (s0 + 32), 8192u + nb * 4096u, 32, 64, T_, T_, D_);
      __builtin_amdgcn_s_wait_tensorcnt(2);
    } else {
      __builtin_amdgcn_s_wait_tensorcnt(0);
    }
    const _Float16* kb = smem + buf * 2048;
    const _Float16* vb = smem + 4096 + buf * 2048;
    const long kstr = 64, vstr = 32;
#else
    const _Float16* kb = K + (long)s0 * D_;
    const _Float16* vb = V + s0;
    const long kstr = 64, vstr = T_;
#endif

    // ---- S = Q K^T : 32 rows x 32 cols ----
    v8f s[2][2] = {{zero, zero}, {zero, zero}};
#pragma unroll
    for (int sub = 0; sub < 2; ++sub) {
      const _Float16* pk = kb + (long)(sub * 16 + cn) * kstr;
      v16h kf0 = load_b_frag(pk, lane);
      v16h kf1 = load_b_frag(pk + 32, lane);
#pragma unroll
      for (int mt = 0; mt < 2; ++mt) {
        s[mt][sub] = WMMA_F16(qf[mt][0], kf0, s[mt][sub]);
        s[mt][sub] = WMMA_F16(qf[mt][1], kf1, s[mt][sub]);
      }
    }

    // ---- scale + causal mask + online softmax ----
#pragma unroll
    for (int mt = 0; mt < 2; ++mt) {
      float rmax[8];
#pragma unroll
      for (int r = 0; r < 8; ++r) {
        const int row = t0 + mt * 16 + r + half8;
#pragma unroll
        for (int sub = 0; sub < 2; ++sub) {
          const int colS = s0 + sub * 16 + cn;
          float v = s[mt][sub][r] * 0.125f;  // 1/sqrt(64)
          if (colS > row) v = -1e30f;
          s[mt][sub][r] = v;
        }
        rmax[r] = fmaxf(s[mt][0][r], s[mt][1][r]);
#pragma unroll
        for (int off = 1; off < 16; off <<= 1)
          rmax[r] = fmaxf(rmax[r], __shfl_xor(rmax[r], off, 32));
      }
#pragma unroll
      for (int r = 0; r < 8; ++r) {
        const float nm = fmaxf(m[mt][r], rmax[r]);
        const float alpha = __expf(m[mt][r] - nm);
        m[mt][r] = nm;
        const float p0 = __expf(s[mt][0][r] - nm);
        const float p1 = __expf(s[mt][1][r] - nm);
        s[mt][0][r] = p0; s[mt][1][r] = p1;
        float rs = p0 + p1;
#pragma unroll
        for (int off = 1; off < 16; off <<= 1)
          rs += __shfl_xor(rs, off, 32);
        l[mt][r] = l[mt][r] * alpha + rs;
#pragma unroll
        for (int nt = 0; nt < 4; ++nt) o[mt][nt][r] *= alpha;
      }
    }

    // ---- P: C-layout f32 -> A-layout f16 via LDS ----
    __syncthreads();
#pragma unroll
    for (int mt = 0; mt < 2; ++mt)
#pragma unroll
      for (int sub = 0; sub < 2; ++sub)
#pragma unroll
        for (int r = 0; r < 8; ++r)
          pt[(mt * 16 + r + half8) * 32 + sub * 16 + cn] = (_Float16)s[mt][sub][r];
    __syncthreads();

    v16h pf[2];
#pragma unroll
    for (int mt = 0; mt < 2; ++mt) {
      const _Float16* pp = pt + (mt * 16 + cn) * 32;
      v8h lo = *(const v8h*)(pp + half8);
      v8h hi = *(const v8h*)(pp + 16 + half8);
#pragma unroll
      for (int i = 0; i < 8; ++i) { pf[mt][i] = lo[i]; pf[mt][i + 8] = hi[i]; }
    }

    // ---- O += P @ V ----
#pragma unroll
    for (int nt = 0; nt < 4; ++nt) {
      const _Float16* pv = vb + (long)(nt * 16 + cn) * vstr;
      v16h vf = load_b_frag(pv, lane);
#pragma unroll
      for (int mt = 0; mt < 2; ++mt)
        o[mt][nt] = WMMA_F16(pf[mt], vf, o[mt][nt]);
    }
  }

  // ---- normalize, store to yh [B,T,C] f16 ----
  const int b = bh / H_, h = bh % H_;
#pragma unroll
  for (int mt = 0; mt < 2; ++mt) {
#pragma unroll
    for (int r = 0; r < 8; ++r) {
      const float inv = 1.0f / l[mt][r];
      const int t = t0 + mt * 16 + r + half8;
#pragma unroll
      for (int nt = 0; nt < 4; ++nt) {
        const int c = h * D_ + nt * 16 + cn;
        yh[((long)b * T_ + t) * C_ + c] = (_Float16)(o[mt][nt][r] * inv);
      }
    }
  }
}

// ---------------------------------------------------------------------------
// Output projection: [BT x C] @ [C x C] + bias -> f32 out
// ---------------------------------------------------------------------------
__global__ __launch_bounds__(256) void gemm_proj(
    const _Float16* __restrict__ yh, const _Float16* __restrict__ wT,
    const float* __restrict__ bias, float* __restrict__ out) {
  const int lane = threadIdx.x & 31;
  const int wave = threadIdx.x >> 5;
  const int m0 = blockIdx.y * 32;
  const int n0 = (blockIdx.x * 8 + wave) * 64;
  const int cn = lane & 15;
  const int half8 = (lane >= 16) ? 8 : 0;

  v8f zero = {0.f, 0.f, 0.f, 0.f, 0.f, 0.f, 0.f, 0.f};
  v8f acc[2][4] = {{zero, zero, zero, zero}, {zero, zero, zero, zero}};

  gemm_core(yh + (long)(m0 + cn) * C_, wT + (long)(n0 + cn) * C_, lane, acc);

#pragma unroll
  for (int nt = 0; nt < 4; ++nt) {
    const int col = n0 + nt * 16 + cn;
    const float bv = bias[col];
#pragma unroll
    for (int mt = 0; mt < 2; ++mt) {
#pragma unroll
      for (int r = 0; r < 8; ++r) {
        const int row = m0 + mt * 16 + r + half8;
        out[(long)row * C_ + col] = acc[mt][nt][r] + bv;
      }
    }
  }
}

// ---------------------------------------------------------------------------
extern "C" void kernel_launch(void* const* d_in, const int* in_sizes, int n_in,
                              void* d_out, int out_size, void* d_ws, size_t ws_size,
                              hipStream_t stream) {
  (void)in_sizes; (void)n_in; (void)out_size; (void)ws_size;
  const float* x      = (const float*)d_in[0];
  const float* w_attn = (const float*)d_in[1];
  const float* b_attn = (const float*)d_in[2];
  const float* w_proj = (const float*)d_in[3];
  const float* b_proj = (const float*)d_in[4];
  float* out = (float*)d_out;

  char* ws = (char*)d_ws;
  const size_t MB = 1024 * 1024;
  _Float16* xh  = (_Float16*)(ws);              // 16 MB  x as f16
  _Float16* waT = (_Float16*)(ws + 16 * MB);    //  6 MB  w_attn^T f16
  _Float16* wpT = (_Float16*)(ws + 22 * MB);    //  2 MB  w_proj^T f16
  _Float16* qh  = (_Float16*)(ws + 24 * MB);    // 16 MB  Q [B,H,T,D]
  _Float16* kh  = (_Float16*)(ws + 40 * MB);    // 16 MB  K [B,H,T,D]
  _Float16* vt  = (_Float16*)(ws + 56 * MB);    // 16 MB  V [B,H,D,T]
  _Float16* yh  = (_Float16*)(ws + 72 * MB);    // 16 MB  attn out [B,T,C]

  cvt_f32_to_f16<<<(BT * C_ / 4 + 255) / 256, 256, 0, stream>>>(x, xh, BT * C_ / 4);
  cvt_transpose<<<((long)C_ * N_QKV + 255) / 256, 256, 0, stream>>>(w_attn, waT, C_, N_QKV);
  cvt_transpose<<<((long)C_ * C_ + 255) / 256, 256, 0, stream>>>(w_proj, wpT, C_, C_);

  gemm_qkv<<<dim3(N_QKV / 512, BT / 32), 256, 0, stream>>>(xh, waT, b_attn, qh, kh, vt);
  attn_fwd<<<dim3(B_ * H_, T_ / 32), 32, 0, stream>>>(qh, kh, vt, yh);
  gemm_proj<<<dim3(C_ / 512, BT / 32), 256, 0, stream>>>(yh, wpT, b_proj, out);
}
